// MHSA_53867479826571
// MI455X (gfx1250) — matrix operand order repaired and checked by reading, not compiled, not run on hardware
//
#include <hip/hip_runtime.h>
#include <hip/hip_bf16.h>
#include <stdint.h>

#define HEADS 12
#define CDIM  768
#define NTOK  1024
#define DH    64

typedef __attribute__((ext_vector_type(16))) __bf16 bf16x16;
typedef __attribute__((ext_vector_type(8)))  float  floatx8;

union FragU {
    uint4   q[2];   // two 16-byte LDS loads
    bf16x16 v;      // WMMA operand view (16 bf16 = 8 VGPRs)
};

__device__ __forceinline__ uint16_t f32_to_bf16(float f) {
    uint32_t u = __float_as_uint(f);
    u += 0x7FFFu + ((u >> 16) & 1u);      // round-to-nearest-even
    return (uint16_t)(u >> 16);
}

__device__ __forceinline__ floatx8 fzero8() {
    floatx8 z;
#pragma unroll
    for (int i = 0; i < 8; ++i) z[i] = 0.0f;
    return z;
}

// --- gfx1250 async global->LDS copy (ASYNCcnt-tracked, no VGPR round-trip) --
__device__ __forceinline__ uint32_t lds_off(const void* p) {
    // flat shared addresses carry the LDS offset in the low 32 bits
    return (uint32_t)(uintptr_t)p;
}
__device__ __forceinline__ void async_copy_b128(uint32_t dst_lds,
                                                const void* src_global) {
    asm volatile("global_load_async_to_lds_b128 %0, %1, off"
                 :: "v"(dst_lds), "v"(src_global) : "memory");
}
// wait until <= n async ops outstanding (in-order completion)
#define WAIT_ASYNC(n) asm volatile("s_wait_asynccnt " #n ::: "memory")

// ---------------------------------------------------------------------------
// x [b][768][1024] f32  ->  xT [b][1024][768] bf16 (token-major), LDS tiled
// ---------------------------------------------------------------------------
__global__ __launch_bounds__(256)
void cvt_transpose_x(const float* __restrict__ x, uint16_t* __restrict__ xT) {
    __shared__ float tile[32][33];
    const int tx = threadIdx.x & 31, ty = threadIdx.x >> 5;
    const int n0 = blockIdx.x * 32, c0 = blockIdx.y * 32, b = blockIdx.z;
    const float* xb = x + (size_t)b * CDIM * NTOK;
    uint16_t* ob = xT + (size_t)b * NTOK * CDIM;
#pragma unroll
    for (int r = 0; r < 4; ++r)
        tile[ty + r * 8][tx] = xb[(size_t)(c0 + ty + r * 8) * NTOK + n0 + tx];
    __syncthreads();
#pragma unroll
    for (int r = 0; r < 4; ++r)
        ob[(size_t)(n0 + ty + r * 8) * CDIM + c0 + tx] =
            f32_to_bf16(tile[tx][ty + r * 8]);
}

// ---------------------------------------------------------------------------
// plain f32 -> bf16 converter (weights)
// ---------------------------------------------------------------------------
__global__ void cvt_f32_bf16(const float* __restrict__ in,
                             uint16_t* __restrict__ out, int n) {
    int i = (blockIdx.x * blockDim.x + threadIdx.x) * 4;
    if (i + 3 < n) {
        float4 f = *(const float4*)(in + i);
        uint2 p;
        p.x = (uint32_t)f32_to_bf16(f.x) | ((uint32_t)f32_to_bf16(f.y) << 16);
        p.y = (uint32_t)f32_to_bf16(f.z) | ((uint32_t)f32_to_bf16(f.w) << 16);
        *(uint2*)(out + i) = p;
    }
}

// ---------------------------------------------------------------------------
// bf16 WMMA GEMM, double-buffered async-to-LDS staging, K-step 64.
//   A : [M][K] bf16 row-major (weights, shared across batch)
//   BT: [N][K] bf16 token-major (activations), batch stride strideB
//   C : OUT_MODE==0 -> CT [N][M] bf16 (token-major, b128 packed stores)
//       OUT_MODE==1 -> C  [M][N] f32
// 256 thr = 8 waves; block tile 128x128; wave tile 32x64.
// ---------------------------------------------------------------------------
template <int OUT_MODE>
__global__ __launch_bounds__(256)
void gemm_bf16_wmma(const uint16_t* __restrict__ A,
                    const uint16_t* __restrict__ BT,
                    void* __restrict__ C,
                    int M, int N, int K,
                    long long strideB, long long strideC) {
    const int KP = 72;  // 144 B rows: multiple of 16 B -> async b128 aligned
    __shared__ __align__(16) uint16_t sA[2][128 * KP];   // [m][k], 64-wide K
    __shared__ __align__(16) uint16_t sB[2][128 * KP];   // [n][k]

    const int tid  = threadIdx.x;
    const int lane = tid & 31;
    const int wid  = tid >> 5;
    const int wm   = wid >> 1;            // 0..3 -> 32-row strip
    const int wn   = wid & 1;             // 0..1 -> 64-col strip
    const int n0   = blockIdx.x * 128;
    const int m0   = blockIdx.y * 128;
    const int b    = blockIdx.z;

    const uint16_t* BTb = BT + (long long)b * strideB;

    floatx8 acc[2][4];
#pragma unroll
    for (int mi = 0; mi < 2; ++mi)
#pragma unroll
        for (int nj = 0; nj < 4; ++nj) acc[mi][nj] = fzero8();

    const int arow = lane & 15;
    const int aklo = (lane >> 4) * 8;     // A-operand K base is 2*aklo (k = 16h + s)
    const int bklo = (lane >> 4) * 16;    // B-operand K base

    // issue one 128x64 A tile + 128x64 B tile: 8 async b128 per thread
    auto stage = [&](int buf, int k0) {
#pragma unroll
        for (int it = 0; it < 4; ++it) {
            int c  = tid + it * 256;          // 1024 chunks of 16 B per tile
            int r  = c >> 3;
            int kq = (c & 7) * 8;
            async_copy_b128(lds_off(&sA[buf][r * KP + kq]),
                            A + (size_t)(m0 + r) * K + k0 + kq);
            async_copy_b128(lds_off(&sB[buf][r * KP + kq]),
                            BTb + (size_t)(n0 + r) * K + k0 + kq);
        }
    };

    const int NST = K / 64;               // 12 pipeline stages
    stage(0, 0);
    for (int s = 0; s < NST; ++s) {
        if (s + 1 < NST) {                // prefetch next tile, then drain
            stage((s + 1) & 1, (s + 1) * 64);   // current one (in-order:
            WAIT_ASYNC(8);                      // <=8 left == newest stage)
        } else {
            WAIT_ASYNC(0);
        }
        __syncthreads();

        const uint16_t* bA = sA[s & 1];
        const uint16_t* bB = sB[s & 1];
#pragma unroll
        for (int ks = 0; ks < 2; ++ks) {  // two 32-deep WMMA sub-steps
            FragU fa[2];
#pragma unroll
            for (int mi = 0; mi < 2; ++mi) {
                const uint16_t* p =
                    &bA[(wm * 32 + mi * 16 + arow) * KP + ks * 32];
                fa[mi].q[0] = *(const uint4*)(p + 2 * aklo);
                fa[mi].q[1] = *(const uint4*)(p + 2 * aklo + 8);
            }
#pragma unroll
            for (int nj = 0; nj < 4; ++nj) {
                FragU fb;
                const uint16_t* p =
                    &bB[(wn * 64 + nj * 16 + arow) * KP + ks * 32 + bklo];
                fb.q[0] = *(const uint4*)(p);
                fb.q[1] = *(const uint4*)(p + 8);
#pragma unroll
                for (int mi = 0; mi < 2; ++mi) {
                    acc[mi][nj] = __builtin_amdgcn_wmma_f32_16x16x32_bf16(
                        false, fa[mi].v, false, fb.v, (short)0, acc[mi][nj],
                        false, false);
                }
            }
        }
        __syncthreads();
    }

    // store (C/D layout: element e -> row e | e+8, column = lane&15)
#pragma unroll
    for (int mi = 0; mi < 2; ++mi) {
#pragma unroll
        for (int nj = 0; nj < 4; ++nj) {
            int mbase = m0 + wm * 32 + mi * 16 + (lane >> 4) * 8;
            int n     = n0 + wn * 64 + nj * 16 + (lane & 15);
            if (OUT_MODE == 0) {
                // token-major bf16: 8 consecutive m per lane -> one b128 store
                uint4 pk;
                pk.x = (uint32_t)f32_to_bf16(acc[mi][nj][0]) |
                       ((uint32_t)f32_to_bf16(acc[mi][nj][1]) << 16);
                pk.y = (uint32_t)f32_to_bf16(acc[mi][nj][2]) |
                       ((uint32_t)f32_to_bf16(acc[mi][nj][3]) << 16);
                pk.z = (uint32_t)f32_to_bf16(acc[mi][nj][4]) |
                       ((uint32_t)f32_to_bf16(acc[mi][nj][5]) << 16);
                pk.w = (uint32_t)f32_to_bf16(acc[mi][nj][6]) |
                       ((uint32_t)f32_to_bf16(acc[mi][nj][7]) << 16);
                *(uint4*)(&((uint16_t*)C)[strideC * b + (size_t)n * M + mbase]) = pk;
            } else {
#pragma unroll
                for (int e = 0; e < 8; ++e)
                    ((float*)C)[strideC * b + (long long)(mbase + e) * N + n] =
                        acc[mi][nj][e];
            }
        }
    }
}

// ---------------------------------------------------------------------------
// Flash attention, double-buffered K (async) / V (register-pipelined).
// qkvT: [B][1024][2304] bf16 token-major (o = s*768+h*64+d).
// ctxT: [B][1024][768] bf16 token-major. One block per (128 queries, h, b).
// ---------------------------------------------------------------------------
__global__ __launch_bounds__(256)
void flash_attn(const uint16_t* __restrict__ qkvT, uint16_t* __restrict__ ctxT) {
    const int DP = 72;  // 144 B rows: multiple of 16 B -> async b128 aligned
    __shared__ __align__(16) uint16_t sQ[128 * DP];        // [i][d]
    __shared__ __align__(16) uint16_t sK[2][64 * DP];      // [j][d]
    __shared__ __align__(16) uint16_t sV[2][64 * DP];      // [d][j]
    __shared__ __align__(16) uint16_t sP[8 * 16 * DP];     // per-wave [row][j]

    const int tid = threadIdx.x, lane = tid & 31, wid = tid >> 5;
    const int n0 = blockIdx.x * 128;
    const int h  = blockIdx.y;
    const int b  = blockIdx.z;
    const int QO = 3 * CDIM;                               // 2304 row pitch

    const uint16_t* base = qkvT + (size_t)b * NTOK * QO;

    // stage Q tile: sQ[i][d] = qkvT[n0+i][h*64+d] -- contiguous rows -> async
#pragma unroll
    for (int it = 0; it < 2; ++it) {
        int c  = tid + it * 256;                  // 512 chunks
        int i  = c >> 2;
        int dq = (c & 3) * 8;
        async_copy_b128(lds_off(&sQ[i * DP + dq]),
                        base + (size_t)(n0 + i) * QO + h * DH + dq);
    }

    // K tile: pure async copy (1 chunk per thread)
    auto stage_k = [&](int buf, int m0) {
        int j  = tid >> 2;
        int dq = (tid & 3) * 8;
        async_copy_b128(lds_off(&sK[buf][j * DP + dq]),
                        base + (size_t)(m0 + j) * QO + CDIM + h * DH + dq);
    };
    // V tile transpose, register-pipelined: loads up front, stores post-compute
    auto load_v = [&](int m0, uint32_t* vr) {
#pragma unroll
        for (int it = 0; it < 8; ++it) {
            int c = tid + it * 256;               // 2048 dword chunks
            int j = c >> 5, d2 = (c & 31) * 2;
            vr[it] = *(const uint32_t*)(
                base + (size_t)(m0 + j) * QO + 2 * CDIM + h * DH + d2);
        }
    };
    auto store_v = [&](int buf, const uint32_t* vr) {
#pragma unroll
        for (int it = 0; it < 8; ++it) {
            int c = tid + it * 256;
            int j = c >> 5, d2 = (c & 31) * 2;
            sV[buf][(d2    ) * DP + j] = (uint16_t)(vr[it] & 0xFFFFu);
            sV[buf][(d2 + 1) * DP + j] = (uint16_t)(vr[it] >> 16);
        }
    };

    const int arow = lane & 15;
    const int aklo = (lane >> 4) * 8;
    const int bklo = (lane >> 4) * 16;
    const int qr0  = wid * 16;

    floatx8 oacc[4];
    float mrow[8], lrow[8];
#pragma unroll
    for (int dt = 0; dt < 4; ++dt) oacc[dt] = fzero8();
#pragma unroll
    for (int e = 0; e < 8; ++e) { mrow[e] = -1e30f; lrow[e] = 0.0f; }

    uint16_t* sPw = &sP[wid * 16 * DP];

    // pipeline prologue: first K/V tile
    {
        uint32_t v0[8];
        stage_k(0, 0);
        load_v(0, v0);
        store_v(0, v0);
    }

    for (int s = 0; s < NTOK / 64; ++s) {
        uint32_t vnext[8];
        if (s + 1 < NTOK / 64) {          // prefetch next tile over compute
            stage_k((s + 1) & 1, (s + 1) * 64);
            load_v((s + 1) * 64, vnext);
            WAIT_ASYNC(1);                // in-order: drains Q + current K
        } else {
            WAIT_ASYNC(0);
        }
        __syncthreads();

        const uint16_t* bK = sK[s & 1];
        const uint16_t* bV = sV[s & 1];

        // S(16x64) = Q_tile x K_tile^T  (dh=64 -> 2 WMMA K-steps)
        FragU fq[2];
#pragma unroll
        for (int h2 = 0; h2 < 2; ++h2) {
            const uint16_t* p = &sQ[(qr0 + arow) * DP + h2 * 32];
            fq[h2].q[0] = *(const uint4*)(p + 2 * aklo);
            fq[h2].q[1] = *(const uint4*)(p + 2 * aklo + 8);
        }
        floatx8 sacc[4];
#pragma unroll
        for (int j = 0; j < 4; ++j) sacc[j] = fzero8();
#pragma unroll
        for (int j = 0; j < 4; ++j) {
#pragma unroll
            for (int h2 = 0; h2 < 2; ++h2) {
                FragU fk;
                const uint16_t* p = &bK[(j * 16 + arow) * DP + h2 * 32 + bklo];
                fk.q[0] = *(const uint4*)(p);
                fk.q[1] = *(const uint4*)(p + 8);
                sacc[j] = __builtin_amdgcn_wmma_f32_16x16x32_bf16(
                    false, fq[h2].v, false, fk.v, (short)0, sacc[j],
                    false, false);
            }
        }

        // scale 1/sqrt(64) + online softmax (xor masks 8..1 stay in-half,
        // matching the C-layout row mapping: element e -> row e | e+8)
#pragma unroll
        for (int j = 0; j < 4; ++j)
#pragma unroll
            for (int e = 0; e < 8; ++e) sacc[j][e] *= 0.125f;

#pragma unroll
        for (int e = 0; e < 8; ++e) {
            float t = fmaxf(fmaxf(sacc[0][e], sacc[1][e]),
                            fmaxf(sacc[2][e], sacc[3][e]));
#pragma unroll
            for (int msk = 8; msk >= 1; msk >>= 1)
                t = fmaxf(t, __shfl_xor(t, msk, 32));
            float mnew = fmaxf(mrow[e], t);
            float corr = __expf(mrow[e] - mnew);
            mrow[e] = mnew;
            float psum = 0.0f;
#pragma unroll
            for (int j = 0; j < 4; ++j) {
                float p = __expf(sacc[j][e] - mnew);
                sacc[j][e] = p;
                psum += p;
            }
#pragma unroll
            for (int msk = 8; msk >= 1; msk >>= 1)
                psum += __shfl_xor(psum, msk, 32);
            lrow[e] = lrow[e] * corr + psum;
#pragma unroll
            for (int dt = 0; dt < 4; ++dt) oacc[dt][e] *= corr;
        }

        // P -> bf16 via per-wave LDS scratch (same-wave DS ops stay in order)
        {
            int prow = (lane >> 4) * 8;
#pragma unroll
            for (int j = 0; j < 4; ++j)
#pragma unroll
                for (int e = 0; e < 8; ++e)
                    sPw[(prow + e) * DP + j * 16 + (lane & 15)] =
                        f32_to_bf16(sacc[j][e]);
        }

        // O(16x64) += P(16x64) x V_tile^T  (64 keys -> 2 WMMA K-steps)
        FragU fp[2];
#pragma unroll
        for (int h2 = 0; h2 < 2; ++h2) {
            const uint16_t* p = &sPw[arow * DP + h2 * 32];
            fp[h2].q[0] = *(const uint4*)(p + 2 * aklo);
            fp[h2].q[1] = *(const uint4*)(p + 2 * aklo + 8);
        }
#pragma unroll
        for (int dt = 0; dt < 4; ++dt) {
#pragma unroll
            for (int h2 = 0; h2 < 2; ++h2) {
                FragU fv;
                const uint16_t* p = &bV[(dt * 16 + arow) * DP + h2 * 32 + bklo];
                fv.q[0] = *(const uint4*)(p);
                fv.q[1] = *(const uint4*)(p + 8);
                oacc[dt] = __builtin_amdgcn_wmma_f32_16x16x32_bf16(
                    false, fp[h2].v, false, fv.v, (short)0, oacc[dt],
                    false, false);
            }
        }

        // transposing stores of the prefetched V tile (latency already hidden)
        if (s + 1 < NTOK / 64) store_v((s + 1) & 1, vnext);
        __syncthreads();
    }

    // normalize by 1/l; write token-major context (lane-coalesced b16 stores)
    uint16_t* outp = ctxT + (size_t)b * NTOK * CDIM;
#pragma unroll
    for (int e = 0; e < 8; ++e) {
        float inv = 1.0f / lrow[e];
        int n = n0 + qr0 + (lane >> 4) * 8 + e;
#pragma unroll
        for (int dt = 0; dt < 4; ++dt) {
            int d = dt * 16 + (lane & 15);
            outp[(size_t)n * CDIM + h * DH + d] = f32_to_bf16(oacc[dt][e] * inv);
        }
    }
}

// ---------------------------------------------------------------------------
// Host-side orchestration
// ---------------------------------------------------------------------------
extern "C" void kernel_launch(void* const* d_in, const int* in_sizes, int n_in,
                              void* d_out, int out_size, void* d_ws,
                              size_t ws_size, hipStream_t stream) {
    (void)in_sizes; (void)n_in; (void)out_size; (void)ws_size;
    const float* x      = (const float*)d_in[0];   // [8][768][32][32]
    const float* w_qkv  = (const float*)d_in[1];   // [2304][768]
    const float* w_proj = (const float*)d_in[2];   // [768][768]
    float* out = (float*)d_out;                    // [8][768][32][32]

    char* ws = (char*)d_ws;
    size_t off = 0;
    uint16_t* xT     = (uint16_t*)(ws + off); off += (size_t)8 * NTOK * CDIM * 2;
    uint16_t* wqkvb  = (uint16_t*)(ws + off); off += (size_t)3 * CDIM * CDIM * 2;
    uint16_t* wprojb = (uint16_t*)(ws + off); off += (size_t)CDIM * CDIM * 2;
    uint16_t* qkvT   = (uint16_t*)(ws + off); off += (size_t)8 * NTOK * 3 * CDIM * 2;
    uint16_t* ctxT   = (uint16_t*)(ws + off);

    // x: transpose + bf16 (token-major); weights: plain bf16
    cvt_transpose_x<<<dim3(NTOK / 32, CDIM / 32, 8), 256, 0, stream>>>(x, xT);
    const int nq = 3 * CDIM * CDIM, np = CDIM * CDIM;
    cvt_f32_bf16<<<nq / 1024, 256, 0, stream>>>(w_qkv, wqkvb, nq);
    cvt_f32_bf16<<<np / 1024, 256, 0, stream>>>(w_proj, wprojb, np);

    // QKV projection: qkvT[b][n][o] = (w_qkv x x[b])^T, bf16 token-major out
    dim3 g1(NTOK / 128, (3 * CDIM) / 128, 8);
    gemm_bf16_wmma<0><<<g1, 256, 0, stream>>>(
        wqkvb, xT, qkvT, 3 * CDIM, NTOK, CDIM,
        (long long)NTOK * CDIM, (long long)NTOK * 3 * CDIM);

    // Fused attention (online softmax), bf16 token-major context out
    dim3 g2(NTOK / 128, HEADS, 8);
    flash_attn<<<g2, 256, 0, stream>>>(qkvT, ctxT);

    // Output projection: out[b] = w_proj x ctx[b], f32 [768][1024] out
    dim3 g3(NTOK / 128, CDIM / 128, 8);
    gemm_bf16_wmma<1><<<g3, 256, 0, stream>>>(
        wprojb, ctxT, out, CDIM, NTOK, CDIM,
        (long long)NTOK * CDIM, (long long)CDIM * NTOK);
}
